// CausalHiFiGANDecoder_86414741996180
// MI455X (gfx1250) — matrix operand order repaired and checked by reading, not compiled, not run on hardware
//
#include <hip/hip_runtime.h>
#include <math.h>

typedef __attribute__((ext_vector_type(16))) _Float16 v16h;
typedef __attribute__((ext_vector_type(8)))  float    v8f;

#define MAXK 11            // largest conv kernel routed through the WMMA kernel
#define WMAX 180           // 128 + (MAXK-1)*max_dil(5) = 178, padded

// ---------------------------------------------------------------------------
// WMMA causal dilated conv1d:  y[b,co,t] = bias[co] + sum_ci sum_j
//   w[co,ci,j] * slope_relu(x[b,ci,t-(K-1-j)*dil])   (+ addsrc[b,co,t])
//
// Block = 128 threads = 4 waves, one 16-co tile, 128 t per block.
// Per 32-ci chunk: stage input window + all-tap weights into LDS (f16, lrelu
// fused), then the tap loop runs purely out of LDS:
//   one A fragment + two B fragments -> two v_wmma_f32_16x16x32_f16.
// Requires T % 128 == 0 and K <= MAXK (true for every conv in this network).
// ---------------------------------------------------------------------------
__global__ __launch_bounds__(128) void conv_wmma_kernel(
    const float* __restrict__ x, const float* __restrict__ w,
    const float* __restrict__ bias, const float* __restrict__ addsrc,
    float* __restrict__ y,
    int Cin, int Cout, int T, int K, int dil, float inSlope)
{
  __shared__ _Float16 xsh[32][WMAX];        // [ci_local][window col]
  __shared__ _Float16 wsh[MAXK][32][18];    // [tap][ci_local][co_local]

  const int tid  = threadIdx.x;
  const int lane = tid & 31;
  const int r    = lane & 15;               // row/col within 16
  const int hi   = lane >> 4;               // half-wave select
  const int wave = tid >> 5;
  const int t0b  = blockIdx.x * 128;        // block time base
  const int co0  = blockIdx.y * 16;

  const int halo = (K - 1) * dil;
  const int W    = 128 + halo;              // staged window width
  const int base = t0b - halo;              // global t of window col 0

  const size_t bx = (size_t)blockIdx.z * (size_t)Cin  * T;
  const size_t by = (size_t)blockIdx.z * (size_t)Cout * T;

  // C/D layout: c[i] -> M = i + hi*8, N = r
  v8f c0, c1;
#pragma unroll
  for (int i = 0; i < 8; ++i) {
    int m  = co0 + i + hi * 8;
    int mm = (m < Cout) ? m : 0;
    float bv = bias[mm];
    bv = (m < Cout) ? bv : 0.0f;
    c0[i] = bv;
    c1[i] = bv;
  }

  for (int kk = 0; kk < Cin; kk += 32) {
    __syncthreads();                        // protect prior-iteration LDS reads

    // ---- stage input window: wave handles 8 rows, 32-lane coalesced cols ----
#pragma unroll
    for (int rr = 0; rr < 8; ++rr) {
      int row  = wave * 8 + rr;
      int grow = kk + row;
      int crow = (grow < Cin) ? grow : 0;   // clamp, keep address in-bounds
      for (int col = lane; col < W; col += 32) {
        int gt = base + col;
        int ct = (gt >= 0) ? gt : 0;
        float f = x[bx + (size_t)crow * T + ct];       // unconditional load
        f = (grow < Cin && gt >= 0) ? f : 0.0f;        // branch-free zero-fill
        f = (f > 0.0f) ? f : f * inSlope;              // fused leaky-relu
        xsh[row][col] = (_Float16)f;
      }
    }
    // ---- stage all-tap weight tile: 16co x 32ci x K ----
    for (int jj = 0; jj < K; ++jj) {
      for (int e = tid; e < 512; e += 128) {
        int ci = e & 31, co = e >> 5;
        int gco = co0 + co, gci = kk + ci;
        int cco = (gco < Cout) ? gco : 0;
        int cci = (gci < Cin) ? gci : 0;
        float wv = w[((size_t)cco * Cin + (size_t)cci) * K + jj];
        wv = (gco < Cout && gci < Cin) ? wv : 0.0f;
        wsh[jj][ci][co] = (_Float16)wv;
      }
    }
    if (kk + 32 < Cin) {
      int pt = (base >= 0) ? base : 0;
      __builtin_prefetch(&x[bx + (size_t)(kk + 32 + lane) * T + pt], 0, 1);
    }
    __syncthreads();

    // ---- tap loop: pure LDS + WMMA ----
    for (int j = 0; j < K; ++j) {
      // A fragment (16x32 f16): lane r = row M(co); VGPR q -> K(ci) pair
      v16h a;
#pragma unroll
      for (int q = 0; q < 8; ++q) {
        int kb = ((q & 4) << 2) + hi * 8 + ((q & 3) << 1);
        a[2 * q]     = wsh[j][kb][r];
        a[2 * q + 1] = wsh[j][kb + 1][r];
      }
      // B fragments (32x16 f16): lane r = time col; VGPR q -> K(ci) pair
      const int colA = wave * 32 + r + j * dil;   // window col (always in range)
      v16h b0, b1;
#pragma unroll
      for (int q = 0; q < 8; ++q) {
        int ci = ((q & 4) << 2) + hi * 8 + ((q & 3) << 1);
        b0[2 * q]     = xsh[ci][colA];
        b0[2 * q + 1] = xsh[ci + 1][colA];
        b1[2 * q]     = xsh[ci][colA + 16];
        b1[2 * q + 1] = xsh[ci + 1][colA + 16];
      }
      c0 = __builtin_amdgcn_wmma_f32_16x16x32_f16(
               false, a, false, b0, (short)0, c0, false, false);
      c1 = __builtin_amdgcn_wmma_f32_16x16x32_f16(
               false, a, false, b1, (short)0, c1, false, false);
    }
  }

  const int t0 = t0b + wave * 32;
#pragma unroll
  for (int i = 0; i < 8; ++i) {
    int m = co0 + i + hi * 8;
    if (m < Cout) {
      size_t off = by + (size_t)m * T + (size_t)(t0 + r);
      float v0 = c0[i];
      float v1 = c1[i];
      if (addsrc) { v0 += addsrc[off]; v1 += addsrc[off + 16]; }  // fused residual
      y[off]      = v0;
      y[off + 16] = v1;
    }
  }
}

// ---------------------------------------------------------------------------
// Grouped causal transposed conv (cpg=2):  y[oc,t] = b[oc] +
//   sum_{c=0,1} sum_j w[oc*2+c, j, 0] * X(oc*2+c, t-j),
//   X(ci,m) valid iff m>=0, m%stride==0, m/stride<Tin.   Tout = Tin*stride.
// ---------------------------------------------------------------------------
__global__ void upsample_kernel(const float* __restrict__ x, const float* __restrict__ w,
                                const float* __restrict__ bias, float* __restrict__ y,
                                int Cin, int Tin, int K, int stride, int B)
{
  const int Cout = Cin >> 1;
  const long long Tout = (long long)Tin * stride;
  const long long N = (long long)B * Cout * Tout;
  for (long long idx = (long long)blockIdx.x * blockDim.x + threadIdx.x; idx < N;
       idx += (long long)gridDim.x * blockDim.x) {
    long long t = idx % Tout;
    long long rest = idx / Tout;
    int oc = (int)(rest % Cout);
    int b  = (int)(rest / Cout);
    const float* x0 = x + (size_t)b * Cin * Tin + (size_t)(oc * 2) * Tin;
    const float* x1 = x0 + Tin;
    const float* w0 = w + (size_t)(oc * 2) * K;
    const float* w1 = w0 + K;
    float acc = bias[oc];
    for (int j = 0; j < K; ++j) {
      long long m = t - j;
      if (m < 0) break;
      if (m % stride == 0) {
        long long ti = m / stride;
        if (ti < Tin) acc += w0[j] * x0[ti] + w1[j] * x1[ti];
      }
    }
    y[(size_t)b * Cout * Tout + (size_t)oc * Tout + t] = acc;
  }
}

// half-snake: channels [0,ns) -> x + sin(a*x)^2/(a+1e-9); [ns,C) -> lrelu 0.01
__global__ void half_snake_kernel(const float* __restrict__ x, const float* __restrict__ alpha,
                                  float* __restrict__ y, int C, int ns, int T, long long N)
{
  for (long long idx = (long long)blockIdx.x * blockDim.x + threadIdx.x; idx < N;
       idx += (long long)gridDim.x * blockDim.x) {
    int c = (int)((idx / T) % C);
    float v = x[idx];
    if (c < ns) {
      float a = alpha[c];
      float s = sinf(a * v);
      y[idx] = v + s * s / (a + 1e-9f);
    } else {
      y[idx] = (v > 0.0f) ? v : v * 0.01f;
    }
  }
}

// dst = b*dst + a*src   (b==0 path avoids reading poisoned dst)
__global__ void axpby_kernel(float* __restrict__ dst, const float* __restrict__ src,
                             float a, float b, long long N)
{
  for (long long idx = (long long)blockIdx.x * blockDim.x + threadIdx.x; idx < N;
       idx += (long long)gridDim.x * blockDim.x) {
    float d = (b != 0.0f) ? b * dst[idx] : 0.0f;
    dst[idx] = d + a * src[idx];
  }
}

// final 27->1 causal conv (K=3) with clip
__global__ void post_kernel(const float* __restrict__ x, const float* __restrict__ w,
                            const float* __restrict__ bias, float* __restrict__ out,
                            int Cin, int K, long long T, int B)
{
  const long long N = (long long)B * T;
  for (long long idx = (long long)blockIdx.x * blockDim.x + threadIdx.x; idx < N;
       idx += (long long)gridDim.x * blockDim.x) {
    long long t = idx % T;
    int b = (int)(idx / T);
    float acc = bias[0];
    const float* xb = x + (size_t)b * Cin * T;
    for (int ci = 0; ci < Cin; ++ci)
      for (int j = 0; j < K; ++j) {
        long long ti = t - (K - 1 - j);
        if (ti >= 0) acc += w[ci * K + j] * xb[(size_t)ci * T + ti];
      }
    acc = fminf(1.0f, fmaxf(-1.0f, acc));
    out[(size_t)b * T + t] = acc;
  }
}

__global__ void len_kernel(const int* __restrict__ tl, int* __restrict__ out, int B, int up)
{
  int i = threadIdx.x;
  if (i < B) out[i] = tl[i] * up;
}

static inline unsigned gdim(long long n, int b) { return (unsigned)((n + b - 1) / b); }

extern "C" void kernel_launch(void* const* d_in, const int* in_sizes, int n_in,
                              void* d_out, int out_size, void* d_ws, size_t ws_size,
                              hipStream_t stream)
{
  (void)in_sizes; (void)n_in; (void)out_size; (void)ws_size;
  constexpr int B = 2, T0 = 128, BASE = 864, TOTAL_UP = 1764;
  const int rates[5] = {7, 7, 6, 3, 2};
  const int resK[3]  = {3, 7, 11};
  const int resD[3]  = {1, 3, 5};

  const float* xin = (const float*)d_in[0];
  const int*   tl  = (const int*)d_in[1];
  int p = 2;
  auto F = [&]() { return (const float*)d_in[p++]; };

  const float* pre_w = F();
  const float* pre_b = F();

  // 4 f32 ping-pong buffers of B * max(C*T) = 2 * 6,096,384 elements each
  const size_t MAXE = (size_t)B * 6096384;
  float* A   = (float*)d_ws;      // pre-conv out / block-state h
  float* S   = A + MAXE;          // snake out / conv1 temp
  float* X   = S + MAXE;          // res-layer input
  float* ACC = X + MAXE;          // block mean accumulator (becomes next h)

  auto conv = [&](const float* in, const float* ww, const float* bb, const float* add,
                  float* outp, int Cin, int Cout, int T, int K, int dil, float slope) {
    dim3 g((unsigned)(T / 128), (unsigned)((Cout + 15) / 16), (unsigned)B);
    conv_wmma_kernel<<<g, dim3(128), 0, stream>>>(in, ww, bb, add, outp,
                                                  Cin, Cout, T, K, dil, slope);
  };

  // pre conv: 16 -> 864, K=7
  conv(xin, pre_w, pre_b, nullptr, A, 16, BASE, T0, 7, 1, 1.0f);

  const float* h = A;
  int C = BASE, T = T0;
  for (int s = 0; s < 5; ++s) {
    const int rate = rates[s], K = 2 * rate, Co = C / 2, To = T * rate;
    const float* alpha = F();
    const float* up_w  = F();
    const float* up_b  = F();

    long long Ns = (long long)B * C * T;
    half_snake_kernel<<<gdim(Ns, 256), 256, 0, stream>>>(h, alpha, S, C, C / 2, T, Ns);

    long long Nu = (long long)B * Co * To;
    upsample_kernel<<<gdim(Nu, 256), 256, 0, stream>>>(S, up_w, up_b, X, C, T, K, rate, B);

    for (int bI = 0; bI < 3; ++bI) {
      const int kk = resK[bI];
      const float* cur = X;                       // block state starts at res input
      for (int dI = 0; dI < 3; ++dI) {
        const float* w1 = F(); const float* b1 = F();
        const float* w2 = F(); const float* b2 = F();
        // t = conv1(lrelu(h), dil) ; h = h + conv2(lrelu(t), 1)
        conv(cur, w1, b1, nullptr, S, Co, Co, To, kk, resD[dI], 0.1f);
        conv(S,   w2, b2, cur,     A, Co, Co, To, kk, 1,        0.1f);
        cur = A;
      }
      axpby_kernel<<<gdim(Nu, 256), 256, 0, stream>>>(ACC, A, 1.0f / 3.0f,
                                                      bI ? 1.0f : 0.0f, Nu);
    }
    h = ACC; C = Co; T = To;
  }

  const float* post_alpha = F();
  const float* post_w     = F();
  const float* post_b     = F();

  long long Ns = (long long)B * C * T;                         // C=27, T=225792
  half_snake_kernel<<<gdim(Ns, 256), 256, 0, stream>>>(h, post_alpha, S, C, 13, T, Ns);

  long long Nt = (long long)B * T;
  post_kernel<<<gdim(Nt, 256), 256, 0, stream>>>(S, post_w, post_b, (float*)d_out,
                                                 C, 3, (long long)T, B);
  len_kernel<<<1, 32, 0, stream>>>(tl, (int*)d_out + (size_t)B * T, B, TOTAL_UP);
}
